// FastAttention_64871186039107
// MI455X (gfx1250) — compile-verified, matrix operand order checked by reading, not run on hardware
//
#include <hip/hip_runtime.h>
#include <hip/hip_bf16.h>
#include <cstdint>
#include <math.h>

typedef __attribute__((ext_vector_type(16))) __bf16 v16bf;
typedef __attribute__((ext_vector_type(8)))  float  v8f;
typedef __attribute__((ext_vector_type(4)))  __bf16 v4bf;

#define HEADS  16
#define KD     64
#define DMODEL 1024
#define NQKV   3072
#define BATCH  4
#define SEQ    4096
#define ROWS   (BATCH*SEQ)   // 16384

#if defined(__gfx1250__) && __has_builtin(__builtin_amdgcn_global_load_async_to_lds_b128) && __has_builtin(__builtin_amdgcn_s_wait_asynccnt)
#define USE_ASYNC_LDS 1
typedef int v4i_t __attribute__((ext_vector_type(4)));
typedef __attribute__((address_space(1))) v4i_t* gp128;   // global src of b128 async copy
typedef __attribute__((address_space(3))) v4i_t* lp128;   // LDS dst of b128 async copy
#endif

__device__ __forceinline__ __bf16 f2bf(float f) { return (__bf16)f; }
__device__ __forceinline__ float  bf2f(__bf16 h) { return (float)h; }

__device__ __forceinline__ v8f wmma_bf16(v16bf a, v16bf b, v8f c) {
  // (neg_a, A, neg_b, B, c_mod, C, reuse_a, reuse_b)
  return __builtin_amdgcn_wmma_f32_16x16x32_bf16(false, a, false, b, (short)0, c,
                                                 false, false);
}

// ---------------- conversion / prep kernels ----------------

__global__ void __launch_bounds__(256) k_conv_x(const float* __restrict__ X,
                                                __bf16* __restrict__ Xb) {
  size_t i = (size_t)blockIdx.x * blockDim.x + threadIdx.x; // 4 elems per thread
  const float4 f = ((const float4*)X)[i];
  v4bf o;
  o[0] = f2bf(f.x); o[1] = f2bf(f.y); o[2] = f2bf(f.z); o[3] = f2bf(f.w);
  ((v4bf*)Xb)[i] = o;
}

// Wt[n][k] = W{q|k|v}[k][n%1024]  (pre-transposed so each output column is
// K-contiguous: one 32B load per WMMA B-fragment lane)
__global__ void __launch_bounds__(256) k_conv_wqkv(const float* __restrict__ Wq,
                                                   const float* __restrict__ Wk,
                                                   const float* __restrict__ Wv,
                                                   __bf16* __restrict__ Wt) {
  int i = blockIdx.x * 256 + threadIdx.x;   // 0 .. 3072*1024-1
  int k = i & (DMODEL - 1);
  int n = i >> 10;
  const float* src = (n < 1024) ? Wq : (n < 2048 ? Wk : Wv);
  Wt[i] = f2bf(src[(size_t)k * DMODEL + (n & 1023)]);
}

__global__ void __launch_bounds__(256) k_conv_wo(const float* __restrict__ Wo,
                                                 __bf16* __restrict__ Wot) {
  int i = blockIdx.x * 256 + threadIdx.x;   // 0 .. 1024*1024-1
  int k = i & (DMODEL - 1);
  int n = i >> 10;
  Wot[i] = f2bf(Wo[(size_t)k * DMODEL + n]);
}

__global__ void __launch_bounds__(256) k_zero_f32(float* __restrict__ p) {
  p[(size_t)blockIdx.x * 256 + threadIdx.x] = 0.0f;
}

__global__ void __launch_bounds__(256) k_conv_ctx(const float* __restrict__ c,
                                                  __bf16* __restrict__ cb) {
  size_t i = (size_t)blockIdx.x * 256 + threadIdx.x;
  cb[i] = f2bf(c[i]);
}

// ---------------- GEMM 1: QKV = Xb(16384x1024) * Wt^T -> bf16(16384x3072) ----

__global__ void __launch_bounds__(256) k_gemm_qkv(const __bf16* __restrict__ Xb,
                                                  const __bf16* __restrict__ Wt,
                                                  __bf16* __restrict__ QKV) {
  const int lane = threadIdx.x & 31;
  const int wave = threadIdx.x >> 5;
  const int lr = lane & 15, lh = lane >> 4;
  const int M0 = blockIdx.y * 128 + (wave & 1) * 64;
  const int N0 = blockIdx.x * 256 + (wave >> 1) * 64;

  const v8f vz = {0.f,0.f,0.f,0.f,0.f,0.f,0.f,0.f};
  v8f acc[4][4];
  for (int mi = 0; mi < 4; mi++)
    for (int ni = 0; ni < 4; ni++) acc[mi][ni] = vz;

  const __bf16* aBase = Xb + (size_t)(M0 + lr) * DMODEL + lh * 16;
  const __bf16* bBase = Wt + (size_t)(N0 + lr) * DMODEL + lh * 16;

  for (int k0 = 0; k0 < DMODEL; k0 += 32) {
    __builtin_prefetch((const void*)(aBase + k0 + 128), 0, 1);
    __builtin_prefetch((const void*)(bBase + k0 + 128), 0, 1);
    v16bf a[4], b[4];
#pragma unroll
    for (int mi = 0; mi < 4; mi++)
      a[mi] = *(const v16bf*)(aBase + (size_t)mi * 16 * DMODEL + k0);
#pragma unroll
    for (int ni = 0; ni < 4; ni++)
      b[ni] = *(const v16bf*)(bBase + (size_t)ni * 16 * DMODEL + k0);
#pragma unroll
    for (int mi = 0; mi < 4; mi++)
#pragma unroll
      for (int ni = 0; ni < 4; ni++)
        acc[mi][ni] = wmma_bf16(a[mi], b[ni], acc[mi][ni]);
  }
#pragma unroll
  for (int mi = 0; mi < 4; mi++)
#pragma unroll
    for (int ni = 0; ni < 4; ni++)
#pragma unroll
      for (int r = 0; r < 8; r++) {
        int row = M0 + mi * 16 + lh * 8 + r;
        int col = N0 + ni * 16 + lr;
        QKV[(size_t)row * NQKV + col] = f2bf(acc[mi][ni][r]);
      }
}

// ---------------- softmax over sequence axis for the K slice ----------------

__global__ void __launch_bounds__(256) k_softmax(__bf16* __restrict__ QKV) {
  const int tx = threadIdx.x & 63, ty = threadIdx.x >> 6;
  const size_t base = (size_t)blockIdx.y * SEQ * NQKV + DMODEL + blockIdx.x * 64 + tx;

  float m = -3.0e38f, s = 0.f;
  for (int r = ty; r < SEQ; r += 4) {
    float x = bf2f(QKV[base + (size_t)r * NQKV]);
    float nm = fmaxf(m, x);
    s = s * __expf(m - nm) + __expf(x - nm);
    m = nm;
  }
  __shared__ float sm[4][64], ss[4][64];
  sm[ty][tx] = m; ss[ty][tx] = s;
  __syncthreads();
  float M = fmaxf(fmaxf(sm[0][tx], sm[1][tx]), fmaxf(sm[2][tx], sm[3][tx]));
  float S = ss[0][tx] * __expf(sm[0][tx] - M) + ss[1][tx] * __expf(sm[1][tx] - M) +
            ss[2][tx] * __expf(sm[2][tx] - M) + ss[3][tx] * __expf(sm[3][tx] - M);
  float inv = 1.0f / S;
  for (int r = ty; r < SEQ; r += 4) {
    size_t idx = base + (size_t)r * NQKV;
    QKV[idx] = f2bf(__expf(bf2f(QKV[idx]) - M) * inv);
  }
}

// ------- GEMM 2: ctx[b,h,d,e] += sum_n Q[n,d]*Ksm[n,e]  (K-split + atomics) --

__global__ void __launch_bounds__(256) k_gemm_ctx(const __bf16* __restrict__ QKV,
                                                  float* __restrict__ ctx) {
  const int lane = threadIdx.x & 31;
  const int wave = threadIdx.x >> 5;
  const int lr = lane & 15, lh = lane >> 4;
  const int bh = blockIdx.y, b = bh >> 4, h = bh & 15;
  const int nbase = blockIdx.x * 512 + wave * 64;

  __shared__ __bf16 stage[8][32][64];  // 32 KB, per-wave slices

  const v8f vz = {0.f,0.f,0.f,0.f,0.f,0.f,0.f,0.f};
  v8f acc[4][4];
  for (int mi = 0; mi < 4; mi++)
    for (int ni = 0; ni < 4; ni++) acc[mi][ni] = vz;

#if USE_ASYNC_LDS
  // per-lane 16B chunk: row = it*4 + lane/8, byte offset (lane%8)*16 in a 128B row
  const int arow = lane >> 3;
  const int acol = (lane & 7) * 16;
  uint8_t* qkv_nc = (uint8_t*)const_cast<__bf16*>(QKV);
#endif

  for (int kk = 0; kk < 2; kk++) {
    const int n0 = nbase + kk * 32;
    // ---- stage Q tile (32 rows x 64 head-channels = 4KB) ----
#if USE_ASYNC_LDS
    {
      uint8_t* gq = qkv_nc + ((size_t)(b * SEQ + n0) * NQKV + h * 64) * 2;
#pragma unroll
      for (int it = 0; it < 8; it++) {
        int row = it * 4 + arow;
        gp128 src = (gp128)(uintptr_t)(gq + (size_t)row * (NQKV * 2) + acol);
        lp128 dst = (lp128)(uint32_t)(uintptr_t)((uint8_t*)&stage[wave][row][0] + acol);
        __builtin_amdgcn_global_load_async_to_lds_b128(src, dst, 0, 0);
      }
      __builtin_amdgcn_s_wait_asynccnt(0);
    }
#else
    for (int r = 0; r < 32; r++) {
      const uint32_t* src =
          (const uint32_t*)(QKV + (size_t)(b * SEQ + n0 + r) * NQKV + h * 64);
      ((uint32_t*)stage[wave][r])[lane] = src[lane];
    }
#endif
    __syncthreads();
    v16bf a[4];
#pragma unroll
    for (int mi = 0; mi < 4; mi++)
#pragma unroll
      for (int j = 0; j < 16; j++)
        a[mi][j] = stage[wave][lh * 16 + j][mi * 16 + lr];  // transposed gather
    __syncthreads();
    // ---- stage Ksm tile ----
#if USE_ASYNC_LDS
    {
      uint8_t* gk = qkv_nc + ((size_t)(b * SEQ + n0) * NQKV + DMODEL + h * 64) * 2;
#pragma unroll
      for (int it = 0; it < 8; it++) {
        int row = it * 4 + arow;
        gp128 src = (gp128)(uintptr_t)(gk + (size_t)row * (NQKV * 2) + acol);
        lp128 dst = (lp128)(uint32_t)(uintptr_t)((uint8_t*)&stage[wave][row][0] + acol);
        __builtin_amdgcn_global_load_async_to_lds_b128(src, dst, 0, 0);
      }
      __builtin_amdgcn_s_wait_asynccnt(0);
    }
#else
    for (int r = 0; r < 32; r++) {
      const uint32_t* src = (const uint32_t*)(QKV + (size_t)(b * SEQ + n0 + r) * NQKV +
                                              DMODEL + h * 64);
      ((uint32_t*)stage[wave][r])[lane] = src[lane];
    }
#endif
    __syncthreads();
    v16bf bb[4];
#pragma unroll
    for (int ni = 0; ni < 4; ni++)
#pragma unroll
      for (int j = 0; j < 16; j++)
        bb[ni][j] = stage[wave][lh * 16 + j][ni * 16 + lr];
    __syncthreads();
#pragma unroll
    for (int mi = 0; mi < 4; mi++)
#pragma unroll
      for (int ni = 0; ni < 4; ni++)
        acc[mi][ni] = wmma_bf16(a[mi], bb[ni], acc[mi][ni]);
  }
  float* cbase = ctx + (size_t)bh * 4096;
#pragma unroll
  for (int mi = 0; mi < 4; mi++)
#pragma unroll
    for (int ni = 0; ni < 4; ni++)
#pragma unroll
      for (int r = 0; r < 8; r++) {
        int d = mi * 16 + lh * 8 + r;
        int e = ni * 16 + lr;
        atomicAdd(cbase + d * 64 + e, acc[mi][ni][r]);
      }
}

// ---- GEMM 3: attn[n, h*64+d] = sum_e V[n, h*64+e] * ctx[b,h,d,e] ----

__global__ void __launch_bounds__(256) k_gemm_av(const __bf16* __restrict__ QKV,
                                                 const __bf16* __restrict__ ctxb,
                                                 __bf16* __restrict__ attn) {
  const int lane = threadIdx.x & 31;
  const int wave = threadIdx.x >> 5;
  const int lr = lane & 15, lh = lane >> 4;
  const int bh = blockIdx.y, b = bh >> 4, h = bh & 15;
  const int nloc = blockIdx.x * 512 + wave * 64;
  const size_t rowBase = (size_t)b * SEQ + nloc;

  const v8f vz = {0.f,0.f,0.f,0.f,0.f,0.f,0.f,0.f};
  v8f acc[4][4];
  for (int mi = 0; mi < 4; mi++)
    for (int ni = 0; ni < 4; ni++) acc[mi][ni] = vz;

#pragma unroll
  for (int k0 = 0; k0 < 64; k0 += 32) {
    v16bf a[4], bb[4];
#pragma unroll
    for (int mi = 0; mi < 4; mi++)
      a[mi] = *(const v16bf*)(QKV + (rowBase + mi * 16 + lr) * NQKV + 2 * DMODEL +
                              h * 64 + k0 + lh * 16);
#pragma unroll
    for (int ni = 0; ni < 4; ni++)
      bb[ni] = *(const v16bf*)(ctxb + (size_t)bh * 4096 + (ni * 16 + lr) * 64 +
                               k0 + lh * 16);
#pragma unroll
    for (int mi = 0; mi < 4; mi++)
#pragma unroll
      for (int ni = 0; ni < 4; ni++)
        acc[mi][ni] = wmma_bf16(a[mi], bb[ni], acc[mi][ni]);
  }
#pragma unroll
  for (int mi = 0; mi < 4; mi++)
#pragma unroll
    for (int ni = 0; ni < 4; ni++)
#pragma unroll
      for (int r = 0; r < 8; r++) {
        size_t row = rowBase + mi * 16 + lh * 8 + r;
        int col = h * 64 + ni * 16 + lr;
        attn[row * DMODEL + col] = f2bf(acc[mi][ni][r]);
      }
}

// ---------------- GEMM 4: out = attn(16384x1024) * Wo -> f32 ----------------

__global__ void __launch_bounds__(256) k_gemm_out(const __bf16* __restrict__ attn,
                                                  const __bf16* __restrict__ Wot,
                                                  float* __restrict__ out) {
  const int lane = threadIdx.x & 31;
  const int wave = threadIdx.x >> 5;
  const int lr = lane & 15, lh = lane >> 4;
  const int M0 = blockIdx.y * 128 + (wave & 1) * 64;
  const int N0 = blockIdx.x * 256 + (wave >> 1) * 64;

  const v8f vz = {0.f,0.f,0.f,0.f,0.f,0.f,0.f,0.f};
  v8f acc[4][4];
  for (int mi = 0; mi < 4; mi++)
    for (int ni = 0; ni < 4; ni++) acc[mi][ni] = vz;

  const __bf16* aBase = attn + (size_t)(M0 + lr) * DMODEL + lh * 16;
  const __bf16* bBase = Wot + (size_t)(N0 + lr) * DMODEL + lh * 16;

  for (int k0 = 0; k0 < DMODEL; k0 += 32) {
    __builtin_prefetch((const void*)(aBase + k0 + 128), 0, 1);
    v16bf a[4], b[4];
#pragma unroll
    for (int mi = 0; mi < 4; mi++)
      a[mi] = *(const v16bf*)(aBase + (size_t)mi * 16 * DMODEL + k0);
#pragma unroll
    for (int ni = 0; ni < 4; ni++)
      b[ni] = *(const v16bf*)(bBase + (size_t)ni * 16 * DMODEL + k0);
#pragma unroll
    for (int mi = 0; mi < 4; mi++)
#pragma unroll
      for (int ni = 0; ni < 4; ni++)
        acc[mi][ni] = wmma_bf16(a[mi], b[ni], acc[mi][ni]);
  }
#pragma unroll
  for (int mi = 0; mi < 4; mi++)
#pragma unroll
    for (int ni = 0; ni < 4; ni++)
#pragma unroll
      for (int r = 0; r < 8; r++) {
        int row = M0 + mi * 16 + lh * 8 + r;
        int col = N0 + ni * 16 + lr;
        out[(size_t)row * DMODEL + col] = acc[mi][ni][r];
      }
}

// ---------------------------------------------------------------------------

extern "C" void kernel_launch(void* const* d_in, const int* in_sizes, int n_in,
                              void* d_out, int out_size, void* d_ws, size_t ws_size,
                              hipStream_t stream) {
  const float* x  = (const float*)d_in[0];
  const float* Wq = (const float*)d_in[1];
  const float* Wk = (const float*)d_in[2];
  const float* Wv = (const float*)d_in[3];
  const float* Wo = (const float*)d_in[4];
  float* out = (float*)d_out;

  char* ws = (char*)d_ws;
  size_t off = 0;
  auto alloc = [&](size_t bytes) -> void* {
    void* p = ws + off;
    off += (bytes + 255) & ~(size_t)255;
    return p;
  };
  __bf16* Xb   = (__bf16*)alloc((size_t)ROWS * DMODEL * 2);   // 32 MB (reused as attn)
  __bf16* Wt   = (__bf16*)alloc((size_t)NQKV * DMODEL * 2);   //  6 MB
  __bf16* Wot  = (__bf16*)alloc((size_t)DMODEL * DMODEL * 2); //  2 MB
  __bf16* QKV  = (__bf16*)alloc((size_t)ROWS * NQKV * 2);     // 96 MB
  float*  ctx  = (float*)alloc((size_t)BATCH * HEADS * KD * KD * 4); // 1 MB
  __bf16* ctxb = (__bf16*)alloc((size_t)BATCH * HEADS * KD * KD * 2);
  __bf16* attn = Xb;  // Xb is dead after k_gemm_qkv

  k_conv_x   <<<dim3(16384), dim3(256), 0, stream>>>(x, Xb);
  k_conv_wqkv<<<dim3(12288), dim3(256), 0, stream>>>(Wq, Wk, Wv, Wt);
  k_conv_wo  <<<dim3(4096),  dim3(256), 0, stream>>>(Wo, Wot);
  k_zero_f32 <<<dim3(1024),  dim3(256), 0, stream>>>(ctx);
  k_gemm_qkv <<<dim3(12, 128), dim3(256), 0, stream>>>(Xb, Wt, QKV);
  k_softmax  <<<dim3(16, 4),   dim3(256), 0, stream>>>(QKV);
  k_gemm_ctx <<<dim3(8, 64),   dim3(256), 0, stream>>>(QKV, ctx);
  k_conv_ctx <<<dim3(1024),    dim3(256), 0, stream>>>(ctx, ctxb);
  k_gemm_av  <<<dim3(8, 64),   dim3(256), 0, stream>>>(QKV, ctxb, attn);
  k_gemm_out <<<dim3(4, 128),  dim3(256), 0, stream>>>(attn, Wot, out);
}
